// T5EncoderBlock_2422361555275
// MI455X (gfx1250) — matrix operand order repaired and checked by reading, not compiled, not run on hardware
//
#include <hip/hip_runtime.h>

// ---------------- problem constants ----------------
constexpr int Bn  = 2;
constexpr int Sn  = 2048;
constexpr int Dn  = 768;
constexpr int Hn  = 12;
constexpr int HDn = 64;
constexpr int FFn = 3072;
constexpr int Mn  = Bn * Sn;            // 4096 rows
constexpr float LN_EPS = 1e-6f;

// ---------------- WMMA types ----------------
typedef __attribute__((ext_vector_type(16))) __bf16 v16bf;
typedef __attribute__((ext_vector_type(8)))  float  v8f;

struct Frag32B { uint4 lo, hi; };

static __device__ __forceinline__ v16bf ld_frag(const unsigned short* p0,
                                                const unsigned short* p1) {
  Frag32B t;
  t.lo = *(const uint4*)p0;
  t.hi = *(const uint4*)p1;
  return __builtin_bit_cast(v16bf, t);
}

static __device__ __forceinline__ unsigned short f2bf(float f) {
  unsigned u = __builtin_bit_cast(unsigned, f);
  u += 0x7FFFu + ((u >> 16) & 1u);                 // round-to-nearest-even
  return (unsigned short)(u >> 16);
}

static __device__ __forceinline__ uint4 cvt8_bf(float4 f0, float4 f1) {
  __align__(16) unsigned short tmp[8] =
    { f2bf(f0.x), f2bf(f0.y), f2bf(f0.z), f2bf(f0.w),
      f2bf(f1.x), f2bf(f1.y), f2bf(f1.z), f2bf(f1.w) };
  return *(const uint4*)tmp;
}

static __device__ __forceinline__ float gelu_tanh(float x) {
  float x3 = x * x * x;
  return 0.5f * x * (1.0f + tanhf(0.7978845608028654f * (x + 0.044715f * x3)));
}

// ---------------- async global->LDS copy (CDNA5) with portable fallback ----------------
#if defined(__has_builtin)
#  if __has_builtin(__builtin_amdgcn_global_load_async_to_lds_b128) && \
      __has_builtin(__builtin_amdgcn_s_wait_asynccnt)
#    define USE_ASYNC_LDS 1
#  endif
#endif
#ifndef USE_ASYNC_LDS
#  define USE_ASYNC_LDS 0
#endif

typedef int v4i_b128 __attribute__((vector_size(16)));
#define AS1G __attribute__((address_space(1)))
#define AS3L __attribute__((address_space(3)))

static __device__ __forceinline__ void copy16_g2l(const unsigned short* g, unsigned short* l) {
#if USE_ASYNC_LDS
  __builtin_amdgcn_global_load_async_to_lds_b128(
      (AS1G v4i_b128*)(uintptr_t)g,                 // flat global addr == AS1 addr
      (AS3L v4i_b128*)(unsigned)(uintptr_t)l,       // low 32 bits of flat == LDS offset
      0, 0);
#else
  *(uint4*)l = *(const uint4*)g;
#endif
}

#if USE_ASYNC_LDS
#  define WAITA(n) __builtin_amdgcn_s_wait_asynccnt(n)
#else
#  define WAITA(n) ((void)0)
#endif

// ---------------- fp32 -> bf16 convert (flat) ----------------
__global__ __launch_bounds__(256)
void k_f32_to_bf16(const float* __restrict__ in, unsigned short* __restrict__ out, int n) {
  int i = blockIdx.x * 256 + threadIdx.x;
  if (i < n) out[i] = f2bf(in[i]);
}

// ---------------- fp32 [K][N] -> bf16 transposed [N][K] (tiled) ----------------
__global__ __launch_bounds__(256)
void k_cvtT_f32_bf16(const float* __restrict__ in, unsigned short* __restrict__ out,
                     int K, int N) {
  __shared__ unsigned short tile[32][33];
  const int n0 = blockIdx.x * 32;
  const int k0 = blockIdx.y * 32;
  const int tx = threadIdx.x & 31, ty = threadIdx.x >> 5;
#pragma unroll
  for (int r = ty; r < 32; r += 8)
    tile[r][tx] = f2bf(in[(size_t)(k0 + r) * N + n0 + tx]);
  __syncthreads();
#pragma unroll
  for (int r = ty; r < 32; r += 8)
    out[(size_t)(n0 + r) * K + k0 + tx] = tile[tx][r];
}

// ---------------- per-head V transpose: vh[S][HD] -> vT[HD][S] (bf16) ----------------
__global__ __launch_bounds__(256)
void k_tr_v_bf16(const unsigned short* __restrict__ v, unsigned short* __restrict__ vT) {
  __shared__ unsigned short tile[32][33];
  const int bh = blockIdx.z;
  const int b = bh / Hn, h = bh - b * Hn;
  const unsigned short* src = v + (size_t)b * Sn * Dn + (size_t)h * Sn * HDn; // [Sn][HDn]
  unsigned short* dst = vT + (size_t)bh * HDn * Sn;                           // [HDn][Sn]
  const int r0 = blockIdx.x * 32;   // S dim
  const int c0 = blockIdx.y * 32;   // HD dim
  const int tx = threadIdx.x & 31, ty = threadIdx.x >> 5;
#pragma unroll
  for (int r = ty; r < 32; r += 8)
    tile[r][tx] = src[(size_t)(r0 + r) * HDn + c0 + tx];
  __syncthreads();
#pragma unroll
  for (int r = ty; r < 32; r += 8)
    dst[(size_t)(c0 + r) * Sn + r0 + tx] = tile[tx][r];
}

// ---------------- generic bf16 GEMM: C = A[MxK] @ Bt[NxK]^T (+epilogues) ----------------
// 128x64 macro-tile, BK=32, 128 threads (4 waves x 32 rows), double-buffered LDS,
// async global->LDS fills (no VGPR staging across the WMMA clause).
#define GF_REL  1
#define GF_GELU 2

__global__ __launch_bounds__(128, 1)
void k_gemm_bf16(const unsigned short* __restrict__ A,
                 const unsigned short* __restrict__ Bt,
                 const float* __restrict__ bias,
                 const float* __restrict__ resid,
                 unsigned short* __restrict__ outb,
                 float* __restrict__ outf,
                 int M, int N, int K, int flags)
{
  __shared__ __align__(16) unsigned short lsA[2][128 * 32];
  __shared__ __align__(16) unsigned short lsB[2][64 * 32];   // [N][K]
  const int m0 = blockIdx.y * 128;
  const int n0 = blockIdx.x * 64;
  const int t  = threadIdx.x;
  const int wv = t >> 5;
  const int ln = t & 31;
  const int hf = ln >> 4;
  const int nl = ln & 15;

  v8f acc[2][4] = {};

  auto fill = [&](int bb, int kk) {
#pragma unroll
    for (int it = 0; it < 4; ++it) {
      int e = (t + it * 128) * 8;
      int r = e >> 5, c = e & 31;
      copy16_g2l(&A[(size_t)(m0 + r) * K + kk + c], &lsA[bb][e]);
    }
#pragma unroll
    for (int it = 0; it < 2; ++it) {
      int e = (t + it * 128) * 8;
      int nr = e >> 5, kc = e & 31;
      copy16_g2l(&Bt[(size_t)(n0 + nr) * K + kk + kc], &lsB[bb][e]);
    }
  };

  fill(0, 0);                                    // prologue: tile 0 -> buffer 0

  const int nsteps = K >> 5;
  int buf = 0;
  for (int step = 0; step < nsteps; ++step) {
    const bool more = (step + 1) < nsteps;
    __syncthreads();                             // prior readers of buf^1 are done
    if (more) fill(buf ^ 1, (step + 1) << 5);    // async fills overlap compute below
    if (more) { WAITA(6); } else { WAITA(0); }   // tile for `buf` complete (in-order)
    __syncthreads();                             // publish `buf` across waves

    v16bf bfr[4];
#pragma unroll
    for (int c4 = 0; c4 < 4; ++c4) {
      const unsigned short* pb = &lsB[buf][(c4 * 16 + nl) * 32 + hf * 16];
      bfr[c4] = ld_frag(pb, pb + 8);
    }
#pragma unroll
    for (int s = 0; s < 2; ++s) {
      const unsigned short* pa = &lsA[buf][(wv * 32 + s * 16 + nl) * 32 + hf * 16];
      v16bf af = ld_frag(pa, pa + 8);
#pragma unroll
      for (int c4 = 0; c4 < 4; ++c4)
        acc[s][c4] = __builtin_amdgcn_wmma_f32_16x16x32_bf16(
            false, af, false, bfr[c4], (short)0, acc[s][c4], false, false);
    }
    buf ^= 1;
  }

#pragma unroll
  for (int s = 0; s < 2; ++s) {
#pragma unroll
    for (int c4 = 0; c4 < 4; ++c4) {
      int n = n0 + c4 * 16 + nl;
      float bn = bias ? bias[n] : 0.0f;
#pragma unroll
      for (int j = 0; j < 8; ++j) {
        int m = m0 + wv * 32 + s * 16 + hf * 8 + j;
        float v = acc[s][c4][j] + bn;
        if (flags & GF_REL) {                      // clip(n - (m mod S), -3, 3)
          int p = n - (m & (Sn - 1));
          p = p < -3 ? -3 : (p > 3 ? 3 : p);
          v += (float)p;
        }
        if (flags & GF_GELU) v = gelu_tanh(v);
        size_t o = (size_t)m * N + n;
        if (resid) v += resid[o];
        if (outf) outf[o] = v;
        if (outb) outb[o] = f2bf(v);
      }
    }
  }
}

// ---------------- attention scores: (q @ k^T) * mask / sqrt(D) ----------------
// 128x64 output tile per block; whole K=64 kept resident in LDS.
__global__ __launch_bounds__(128, 1)
void k_attn_scores(const unsigned short* __restrict__ qb,
                   const unsigned short* __restrict__ kb,
                   const float* __restrict__ mask,
                   float* __restrict__ attn)
{
  __shared__ __align__(16) unsigned short lsQ[128 * 64];
  __shared__ __align__(16) unsigned short lsK[64 * 64];
  const int j0 = blockIdx.x * 64;
  const int i0 = blockIdx.y * 128;
  const int bh = blockIdx.z;
  const int b  = bh / Hn;
  const int h  = bh - b * Hn;
  const int t = threadIdx.x, wv = t >> 5, ln = t & 31, hf = ln >> 4, nl = ln & 15;

  const size_t headoff = (size_t)b * Sn * Dn + (size_t)h * Sn * HDn;  // raw-reshape head split
  const unsigned short* qt = qb + headoff + (size_t)i0 * HDn;   // contiguous 128x64
  const unsigned short* kt = kb + headoff + (size_t)j0 * HDn;   // contiguous 64x64

#pragma unroll
  for (int it = 0; it < 8; ++it) {
    int e = (t + it * 128) * 8;
    copy16_g2l(&qt[e], &lsQ[e]);
  }
#pragma unroll
  for (int it = 0; it < 4; ++it) {
    int e = (t + it * 128) * 8;
    copy16_g2l(&kt[e], &lsK[e]);
  }
  WAITA(0);
  __syncthreads();

  v8f acc[2][4] = {};
#pragma unroll
  for (int kk = 0; kk < HDn; kk += 32) {
    v16bf bfr[4];
#pragma unroll
    for (int c4 = 0; c4 < 4; ++c4) {
      const unsigned short* pb = &lsK[(c4 * 16 + nl) * 64 + kk + hf * 16];
      bfr[c4] = ld_frag(pb, pb + 8);
    }
#pragma unroll
    for (int s = 0; s < 2; ++s) {
      const unsigned short* pa = &lsQ[(wv * 32 + s * 16 + nl) * 64 + kk + hf * 16];
      v16bf af = ld_frag(pa, pa + 8);
#pragma unroll
      for (int c4 = 0; c4 < 4; ++c4)
        acc[s][c4] = __builtin_amdgcn_wmma_f32_16x16x32_bf16(
            false, af, false, bfr[c4], (short)0, acc[s][c4], false, false);
    }
  }

  const float sc = 0.03608439182435161f;         // 1/sqrt(768)
  float* arow = attn + (size_t)bh * Sn * Sn;
  const float* mrow = mask + (size_t)b * Sn * Sn;
#pragma unroll
  for (int s = 0; s < 2; ++s) {
#pragma unroll
    for (int c4 = 0; c4 < 4; ++c4) {
      int jj = j0 + c4 * 16 + nl;
#pragma unroll
      for (int j = 0; j < 8; ++j) {
        int i = i0 + wv * 32 + s * 16 + hf * 8 + j;
        arow[(size_t)i * Sn + jj] = acc[s][c4][j] * sc * mrow[(size_t)i * Sn + jj];
      }
    }
  }
}

// ---------------- row softmax over attn (in place, row = 2048 fp32) ----------------
__global__ __launch_bounds__(256)
void k_softmax(float* __restrict__ a)
{
  __shared__ float red[256];
  const size_t base = (size_t)blockIdx.x * Sn;
  const int t = threadIdx.x;
  float v[8];
  float m = -3.4e38f;
#pragma unroll
  for (int k = 0; k < 8; ++k) { v[k] = a[base + t + k * 256]; m = fmaxf(m, v[k]); }
  red[t] = m; __syncthreads();
  for (int s = 128; s > 0; s >>= 1) { if (t < s) red[t] = fmaxf(red[t], red[t + s]); __syncthreads(); }
  m = red[0]; __syncthreads();
  float sum = 0.0f;
#pragma unroll
  for (int k = 0; k < 8; ++k) { v[k] = __expf(v[k] - m); sum += v[k]; }
  red[t] = sum; __syncthreads();
  for (int s = 128; s > 0; s >>= 1) { if (t < s) red[t] += red[t + s]; __syncthreads(); }
  float inv = 1.0f / red[0];
#pragma unroll
  for (int k = 0; k < 8; ++k) a[base + t + k * 256] = v[k] * inv;
}

// ---------------- ctx = attn @ v_head ----------------
// 128x64 tile, double-buffered LDS, K = 2048. A: fp32->bf16 converted in the fill
// phase (short register liveness); B: async copy from pre-transposed vT[HD][S].
__global__ __launch_bounds__(128, 1)
void k_attn_ctx(const float* __restrict__ attn,
                const unsigned short* __restrict__ vT,
                unsigned short* __restrict__ ctxb)
{
  __shared__ __align__(16) unsigned short lsA[2][128 * 32];
  __shared__ __align__(16) unsigned short lsB[2][64 * 32];
  const int i0 = blockIdx.x * 128;
  const int bh = blockIdx.y;
  const int b = bh / Hn, h = bh - b * Hn;
  const int t = threadIdx.x, wv = t >> 5, ln = t & 31, hf = ln >> 4, nl = ln & 15;

  const float* at = attn + (size_t)bh * Sn * Sn;
  const unsigned short* vtb = vT + (size_t)bh * HDn * Sn;   // [HD][S]

  v8f acc[2][4] = {};

  auto fillA = [&](int bb, int kk) {             // fp32 -> bf16, synchronous
#pragma unroll
    for (int it = 0; it < 4; ++it) {
      int e = (t + it * 128) * 8;
      int r = e >> 5, c = e & 31;
      const float* src = &at[(size_t)(i0 + r) * Sn + kk + c];
      *(uint4*)&lsA[bb][e] = cvt8_bf(*(const float4*)src, *(const float4*)(src + 4));
    }
  };
  auto fillB = [&](int bb, int kk) {             // pure copy, async when available
#pragma unroll
    for (int it = 0; it < 2; ++it) {
      int e = (t + it * 128) * 8;
      int dr = e >> 5, kc = e & 31;
      copy16_g2l(&vtb[(size_t)dr * Sn + kk + kc], &lsB[bb][e]);
    }
  };

  fillA(0, 0);
  fillB(0, 0);

  const int nsteps = Sn >> 5;                    // 64
  int buf = 0;
  for (int step = 0; step < nsteps; ++step) {
    const bool more = (step + 1) < nsteps;
    __syncthreads();
    if (more) { fillA(buf ^ 1, (step + 1) << 5); fillB(buf ^ 1, (step + 1) << 5); }
    if (more) { WAITA(2); } else { WAITA(0); }
    __syncthreads();

    v16bf bfr[4];
#pragma unroll
    for (int c4 = 0; c4 < 4; ++c4) {
      const unsigned short* pb = &lsB[buf][(c4 * 16 + nl) * 32 + hf * 16];
      bfr[c4] = ld_frag(pb, pb + 8);
    }
#pragma unroll
    for (int s = 0; s < 2; ++s) {
      const unsigned short* pa = &lsA[buf][(wv * 32 + s * 16 + nl) * 32 + hf * 16];
      v16bf af = ld_frag(pa, pa + 8);
#pragma unroll
      for (int c4 = 0; c4 < 4; ++c4)
        acc[s][c4] = __builtin_amdgcn_wmma_f32_16x16x32_bf16(
            false, af, false, bfr[c4], (short)0, acc[s][c4], false, false);
    }
    buf ^= 1;
  }

  unsigned short* ct = ctxb + (size_t)b * Sn * Dn + (size_t)h * Sn * HDn;
#pragma unroll
  for (int s = 0; s < 2; ++s) {
#pragma unroll
    for (int c4 = 0; c4 < 4; ++c4) {
      int n = c4 * 16 + nl;
#pragma unroll
      for (int j = 0; j < 8; ++j) {
        int m = i0 + wv * 32 + s * 16 + hf * 8 + j;
        ct[(size_t)m * HDn + n] = f2bf(acc[s][c4][j]);
      }
    }
  }
}

// ---------------- row LayerNorm (row = 768) ----------------
__global__ __launch_bounds__(256)
void k_layernorm(const float* __restrict__ in, const float* __restrict__ sc,
                 const float* __restrict__ bs, float* __restrict__ outf,
                 unsigned short* __restrict__ outb)
{
  __shared__ float red[256];
  const size_t base = (size_t)blockIdx.x * Dn;
  const int t = threadIdx.x;
  float v0 = in[base + t], v1 = in[base + t + 256], v2 = in[base + t + 512];
  red[t] = v0 + v1 + v2;
  __syncthreads();
  for (int s = 128; s > 0; s >>= 1) { if (t < s) red[t] += red[t + s]; __syncthreads(); }
  float mean = red[0] * (1.0f / Dn);
  __syncthreads();
  float d0 = v0 - mean, d1 = v1 - mean, d2 = v2 - mean;
  red[t] = d0 * d0 + d1 * d1 + d2 * d2;
  __syncthreads();
  for (int s = 128; s > 0; s >>= 1) { if (t < s) red[t] += red[t + s]; __syncthreads(); }
  float rstd = rsqrtf(red[0] * (1.0f / Dn) + LN_EPS);
  float y0 = d0 * rstd * sc[t]       + bs[t];
  float y1 = d1 * rstd * sc[t + 256] + bs[t + 256];
  float y2 = d2 * rstd * sc[t + 512] + bs[t + 512];
  if (outf) { outf[base + t] = y0; outf[base + t + 256] = y1; outf[base + t + 512] = y2; }
  if (outb) { outb[base + t] = f2bf(y0); outb[base + t + 256] = f2bf(y1); outb[base + t + 512] = f2bf(y2); }
}

// ---------------- host-side orchestration ----------------
extern "C" void kernel_launch(void* const* d_in, const int* in_sizes, int n_in,
                              void* d_out, int out_size, void* d_ws, size_t ws_size,
                              hipStream_t stream)
{
  (void)in_sizes; (void)n_in; (void)out_size; (void)ws_size;

  const float* x    = (const float*)d_in[0];
  const float* mask = (const float*)d_in[1];
  const float* Wq = (const float*)d_in[2];  const float* bq = (const float*)d_in[3];
  const float* Wk = (const float*)d_in[4];  const float* bk = (const float*)d_in[5];
  const float* Wv = (const float*)d_in[6];  const float* bv = (const float*)d_in[7];
  const float* Wo = (const float*)d_in[8];  const float* bo = (const float*)d_in[9];
  const float* W1 = (const float*)d_in[10]; const float* b1 = (const float*)d_in[11];
  const float* W2 = (const float*)d_in[12]; const float* b2 = (const float*)d_in[13];
  const float* ln1s = (const float*)d_in[14]; const float* ln1b = (const float*)d_in[15];
  const float* ln2s = (const float*)d_in[16]; const float* ln2b = (const float*)d_in[17];

  float* x2   = (float*)d_out;                               // (B,S,D)
  float* attn = (float*)d_out + (size_t)Bn * Sn * Dn;        // (B,H,S,S)

  char* w = (char*)d_ws;
  size_t off = 0;
  auto alloc = [&](size_t bytes) -> void* {
    void* p = w + off;
    off += (bytes + 255) & ~(size_t)255;
    return p;
  };

  const size_t nXD = (size_t)Bn * Sn * Dn;    // 3,145,728
  const size_t nW  = (size_t)Dn * Dn;         //   589,824
  const size_t nWF = (size_t)Dn * FFn;        // 2,359,296

  unsigned short* xb   = (unsigned short*)alloc(nXD * 2);
  unsigned short* wqb  = (unsigned short*)alloc(nW * 2);     // [N][K] transposed
  unsigned short* wkb  = (unsigned short*)alloc(nW * 2);
  unsigned short* wvb  = (unsigned short*)alloc(nW * 2);
  unsigned short* wob  = (unsigned short*)alloc(nW * 2);
  unsigned short* w1b  = (unsigned short*)alloc(nWF * 2);    // [FF][D]
  unsigned short* w2b  = (unsigned short*)alloc(nWF * 2);    // [D][FF]
  unsigned short* qb   = (unsigned short*)alloc(nXD * 2);
  unsigned short* kb   = (unsigned short*)alloc(nXD * 2);
  unsigned short* vbuf = (unsigned short*)alloc(nXD * 2);
  unsigned short* vT   = (unsigned short*)alloc(nXD * 2);    // [B*H][HD][S]
  unsigned short* ctxb = (unsigned short*)alloc(nXD * 2);
  float*          pre1 = (float*)alloc(nXD * 4);             // also reused as pre2
  float*          x1f  = (float*)alloc(nXD * 4);
  unsigned short* x1b  = (unsigned short*)alloc(nXD * 2);
  unsigned short* ff1b = (unsigned short*)alloc((size_t)Mn * FFn * 2);

  // 1) conversions: x flat; weights converted + transposed to [N][K]
  k_f32_to_bf16<<<dim3((unsigned)((nXD + 255) / 256)), 256, 0, stream>>>(x, xb, (int)nXD);
  k_cvtT_f32_bf16<<<dim3(Dn / 32, Dn / 32),  256, 0, stream>>>(Wq, wqb, Dn, Dn);
  k_cvtT_f32_bf16<<<dim3(Dn / 32, Dn / 32),  256, 0, stream>>>(Wk, wkb, Dn, Dn);
  k_cvtT_f32_bf16<<<dim3(Dn / 32, Dn / 32),  256, 0, stream>>>(Wv, wvb, Dn, Dn);
  k_cvtT_f32_bf16<<<dim3(Dn / 32, Dn / 32),  256, 0, stream>>>(Wo, wob, Dn, Dn);
  k_cvtT_f32_bf16<<<dim3(FFn / 32, Dn / 32), 256, 0, stream>>>(W1, w1b, Dn, FFn);
  k_cvtT_f32_bf16<<<dim3(Dn / 32, FFn / 32), 256, 0, stream>>>(W2, w2b, FFn, Dn);

  // 2) q,k,v projections (rel-bias added to q and v, per reference)
  dim3 gP(Dn / 64, Mn / 128);                                // (12,32)
  k_gemm_bf16<<<gP, 128, 0, stream>>>(xb, wqb, bq, nullptr, qb,   nullptr, Mn, Dn, Dn, GF_REL);
  k_gemm_bf16<<<gP, 128, 0, stream>>>(xb, wkb, bk, nullptr, kb,   nullptr, Mn, Dn, Dn, 0);
  k_gemm_bf16<<<gP, 128, 0, stream>>>(xb, wvb, bv, nullptr, vbuf, nullptr, Mn, Dn, Dn, GF_REL);
  k_tr_v_bf16<<<dim3(Sn / 32, HDn / 32, Bn * Hn), 256, 0, stream>>>(vbuf, vT);

  // 3) attention
  k_attn_scores<<<dim3(Sn / 64, Sn / 128, Bn * Hn), 128, 0, stream>>>(qb, kb, mask, attn);
  k_softmax<<<dim3(Bn * Hn * Sn), 256, 0, stream>>>(attn);
  k_attn_ctx<<<dim3(Sn / 128, Bn * Hn), 128, 0, stream>>>(attn, vT, ctxb);

  // 4) output projection + residual, LN1
  k_gemm_bf16<<<gP, 128, 0, stream>>>(ctxb, wob, bo, x, nullptr, pre1, Mn, Dn, Dn, 0);
  k_layernorm<<<dim3(Mn), 256, 0, stream>>>(pre1, ln1s, ln1b, x1f, x1b);

  // 5) feed-forward (+gelu), +residual, LN2 -> x2
  k_gemm_bf16<<<dim3(FFn / 64, Mn / 128), 128, 0, stream>>>(x1b, w1b, b1, nullptr, ff1b, nullptr, Mn, FFn, Dn, GF_GELU);
  k_gemm_bf16<<<gP, 128, 0, stream>>>(ff1b, w2b, b2, x1f, nullptr, pre1, Mn, Dn, FFn, 0);
  k_layernorm<<<dim3(Mn), 256, 0, stream>>>(pre1, ln2s, ln2b, x2, nullptr);
}
